// IMUCNN1DEncoder_79448305041781
// MI455X (gfx1250) — matrix operand order repaired and checked by reading, not compiled, not run on hardware
//
#include <hip/hip_runtime.h>

typedef __attribute__((ext_vector_type(16))) __bf16 v16bf;
typedef __attribute__((ext_vector_type(8)))  float  v8f;
typedef __attribute__((ext_vector_type(8)))  unsigned int v8u;

// ---------------- workspace layout (dword offsets) ----------------
// A fragments: [kstep][mtile][lane(32)][dword(8)] packed bf16 pairs
// bias fragments: [mtile][lane(32)][r(8)] f32
#define OFF_A1 0        /* 1*4*256   = 1024  */
#define OFF_A2 1024     /* 6*8*256   = 12288 */
#define OFF_A3 13312    /* 12*16*256 = 49152 */
#define OFF_AP 62464    /* 8*16*256  = 32768 */
#define OFF_B1 95232    /* 4*256  */
#define OFF_B2 96256    /* 8*256  */
#define OFF_B3 98304    /* 16*256 */
#define OFF_BP 102400   /* 16*256 */

#define EPS 1e-5f
#define SLOPE 0.1f

// ================= prep: weights -> WMMA A fragments (BN folded) =================
__global__ void prep_A(const float* __restrict__ w, const float* __restrict__ g,
                       const float* __restrict__ v, unsigned int* __restrict__ outA,
                       int C, int MTILES, int KSTEPS, int mode, int total)
{
    int tid = blockIdx.x * blockDim.x + threadIdx.x;
    if (tid >= total) return;
    int j    = tid & 7;
    int lane = (tid >> 3) & 31;
    int mt   = (tid >> 8) % MTILES;
    int s    = (tid >> 8) / MTILES;
    int m    = mt * 16 + (lane & 15);         // A row M = lane%16 (ISA A layout)
    int half = lane >> 4;
    int k0   = 2 * j + 16 * half;             // contiguous K layout: k = 16*half + element (matches B)
    int kg   = s * 32 + k0;
    float f0 = 0.f, f1 = 0.f;
    if (mode == 0) {                          // conv, fold BN scale
        float sc = g[m] * rsqrtf(v[m] + EPS);
        int Ktot = 3 * C;                     // k = dk*C + i
        if (kg < Ktot)     f0 = w[(m * C + (kg % C)) * 3 + (kg / C)] * sc;
        if (kg + 1 < Ktot) f1 = w[(m * C + ((kg + 1) % C)) * 3 + ((kg + 1) / C)] * sc;
    } else {                                  // linear: C == K
        f0 = w[m * C + kg];
        f1 = w[m * C + kg + 1];
    }
    union { unsigned int u; __bf16 h[2]; } pk;
    pk.h[0] = (__bf16)f0; pk.h[1] = (__bf16)f1;
    outA[tid] = pk.u;
}

__global__ void prep_bias(const float* __restrict__ b, const float* __restrict__ g,
                          const float* __restrict__ be, const float* __restrict__ m_,
                          const float* __restrict__ v, float* __restrict__ outB,
                          int mode, int total)
{
    int tid = blockIdx.x * blockDim.x + threadIdx.x;
    if (tid >= total) return;
    int r    = tid & 7;
    int lane = (tid >> 3) & 31;
    int mt   = tid >> 8;
    int ch   = mt * 16 + r + 8 * (lane >> 4); // C/D layout: M = r + 8*(lane>=16)
    float val;
    if (mode == 0) {
        float sc = g[ch] * rsqrtf(v[ch] + EPS);
        val = sc * (b[ch] - m_[ch]) + be[ch];
    } else {
        val = b[ch];
    }
    outB[tid] = val;
}

// ================= main fused encoder =================
struct alignas(32) WaveLds {
    float  xbuf[2][18 * 6];     // zero-padded input [seg-pair][l+1][c] f32
    __bf16 a1[2][18 * 64];      // conv1 out, zero-padded rows 0,17
    __bf16 a2[2][18 * 128];     // conv2 out
    __bf16 pooled[16 * 256];    // [seg][ch] pooled means (bf16) for projection B
};                              // 22880 B per wave (multiple of 32 -> both slices aligned)

__device__ __forceinline__ v16bf loadA_frag(const unsigned int* __restrict__ base,
                                            int fragIdx, int lane)
{
    unsigned off = (unsigned)(fragIdx * 32 + lane) * 8;
    v8u a = *(const v8u*)(base + off);       // -> 2x global_load_b128
    return __builtin_bit_cast(v16bf, a);
}

// B 32x16 bf16: lanes 0-15 hold K 0..15 (dword j = K 2j,2j+1), lanes 16-31 hold K 16..31.
// Key identity: act[(col+dk)*C + i] == act[col*C + kg]  (kg = dk*C + i), so each lane's
// half-window of 16 K-values is CONTIGUOUS: one 32-byte LDS read (2x ds_load_b128).
template<int C>
__device__ __forceinline__ v16bf buildB_win(const __bf16* __restrict__ act, int s, int lane)
{
    int col = lane & 15, half = lane >> 4;
    const v8u* p = (const v8u*)(act + col * C + s * 32 + 16 * half);
    return __builtin_bit_cast(v16bf, *p);
}

__device__ __forceinline__ v16bf buildB_conv1(const float* __restrict__ xb, int lane)
{
    int col = lane & 15, half = lane >> 4;
    union { unsigned int u[8]; v16bf v; } b;
#pragma unroll
    for (int j = 0; j < 8; ++j) {
        int kg  = 2 * j + 16 * half;
        int kgs = kg < 18 ? kg : 0;            // clamp to a safe address (branchless)
        const float* p = &xb[col * 6 + kgs];   // contiguous window, pairs mergeable
        union { unsigned int u; __bf16 h[2]; } t;
        t.h[0] = (__bf16)p[0]; t.h[1] = (__bf16)p[1];
        b.u[j] = kg < 18 ? t.u : 0u;           // K padding 18..31 -> 0
    }
    return b.v;
}

// LAYER: 1 = conv1 (xbuf->a1), 2 = conv2 (a1->a2), 3 = conv3 (a2->pooled), 4 = proj (pooled->out)
template<int LAYER>
__device__ __forceinline__ void wmma_layer(WaveLds& L,
    const unsigned int* __restrict__ Afrag, const float* __restrict__ biasFrag,
    float* __restrict__ outG, int segBase, int si, int lane)
{
    constexpr int KSTEPS = LAYER == 1 ? 1 : LAYER == 2 ? 6  : LAYER == 3 ? 12  : 8;
    constexpr int MTILES = LAYER == 1 ? 4 : LAYER == 2 ? 8  : 16;
    constexpr int NSEG   = LAYER == 4 ? 1 : 2;   // segment pairing amortizes A loads
    constexpr int NM     = 4;                    // acc tiles per pass
    const int half = lane >> 4;
    const int col  = lane & 15;

#pragma unroll 1   // keep passes independent: bounds live VGPRs to one pass
    for (int mb = 0; mb < MTILES; mb += NM) {
        v8f acc[NSEG][NM];
#pragma unroll
        for (int t = 0; t < NM; ++t) {
            v8f bia = *(const v8f*)(biasFrag + (unsigned)((mb + t) * 32 + lane) * 8);
#pragma unroll
            for (int p = 0; p < NSEG; ++p) acc[p][t] = bia;
        }

#pragma unroll 2
        for (int s = 0; s < KSTEPS; ++s) {
            v16bf Bm[NSEG];
#pragma unroll
            for (int p = 0; p < NSEG; ++p) {
                if      constexpr (LAYER == 1) Bm[p] = buildB_conv1(L.xbuf[p], lane);
                else if constexpr (LAYER == 2) Bm[p] = buildB_win<64>(L.a1[p], s, lane);
                else if constexpr (LAYER == 3) Bm[p] = buildB_win<128>(L.a2[p], s, lane);
                else                           Bm[p] = buildB_win<256>(L.pooled, s, lane);
            }
#pragma unroll
            for (int t = 0; t < NM; ++t) {
                v16bf Am = loadA_frag(Afrag, s * MTILES + mb + t, lane);
#pragma unroll
                for (int p = 0; p < NSEG; ++p)
                    acc[p][t] = __builtin_amdgcn_wmma_f32_16x16x32_bf16(
                        false, Am, false, Bm[p], (short)0, acc[p][t], false, false);
            }
        }

        // ---- epilogue ----
#pragma unroll
        for (int p = 0; p < NSEG; ++p) {
#pragma unroll
            for (int t = 0; t < NM; ++t) {
                if constexpr (LAYER <= 2) {
                    __bf16* actOut = (LAYER == 1) ? L.a1[p] : L.a2[p];
#pragma unroll
                    for (int r = 0; r < 8; r += 2) {      // packed u32 bf16 stores
                        float y0 = acc[p][t][r];
                        float y1 = acc[p][t][r + 1];
                        y0 = fmaxf(y0, SLOPE * y0);        // LeakyReLU
                        y1 = fmaxf(y1, SLOPE * y1);
                        union { unsigned int u; __bf16 h[2]; } pk;
                        pk.h[0] = (__bf16)y0; pk.h[1] = (__bf16)y1;
                        int ch = (mb + t) * 16 + r + 8 * half;   // D: M = r + 8*half
                        *(unsigned int*)&actOut[(col + 1) * (MTILES * 16) + ch] = pk.u;
                    }
                } else if constexpr (LAYER == 3) {
#pragma unroll
                    for (int r = 0; r < 8; ++r) {
                        float y = acc[p][t][r];
                        y = fmaxf(y, SLOPE * y);
                        y += __shfl_xor(y, 1, 32);   // sum over N (=L) within 16-lane half
                        y += __shfl_xor(y, 2, 32);
                        y += __shfl_xor(y, 4, 32);
                        y += __shfl_xor(y, 8, 32);
                        int ch = (mb + t) * 16 + r + 8 * half;
                        if (col == 0)
                            L.pooled[(si + p) * 256 + ch] = (__bf16)(y * 0.0625f);
                    }
                } else {
                    // contiguous 8 floats per lane -> 2x global_store_b128
                    *(v8f*)(outG + (size_t)(segBase + col) * 256 + (mb + t) * 16 + 8 * half)
                        = acc[0][t];
                }
            }
        }
    }
}

__launch_bounds__(64, 4)   // min 4 waves/EU => VGPRs capped <= 256, no MSB-toggle mode
__global__ void imu_encoder_main(const float* __restrict__ imu,
                                 const unsigned int* __restrict__ ws,
                                 float* __restrict__ out)
{
    __shared__ WaveLds lds[2];
    const int lane = threadIdx.x & 31;
    const int wave = threadIdx.x >> 5;
    WaveLds& L = lds[wave];

    // zero the L-padding rows (written once; only rows 1..16 touched afterwards)
#pragma unroll
    for (int p = 0; p < 2; ++p) {
        if (lane < 6) { L.xbuf[p][lane] = 0.f; L.xbuf[p][17 * 6 + lane] = 0.f; }
#pragma unroll
        for (int q = 0; q < 2; ++q) {
            int i = q * 32 + lane;
            L.a1[p][i] = (__bf16)0.f; L.a1[p][17 * 64 + i] = (__bf16)0.f;
        }
#pragma unroll
        for (int q = 0; q < 4; ++q) {
            int i = q * 32 + lane;
            L.a2[p][i] = (__bf16)0.f; L.a2[p][17 * 128 + i] = (__bf16)0.f;
        }
    }

    const unsigned int* A1 = ws + OFF_A1;
    const unsigned int* A2 = ws + OFF_A2;
    const unsigned int* A3 = ws + OFF_A3;
    const unsigned int* AP = ws + OFF_AP;
    const float* B1 = (const float*)(ws + OFF_B1);
    const float* B2 = (const float*)(ws + OFF_B2);
    const float* B3 = (const float*)(ws + OFF_B3);
    const float* BP = (const float*)(ws + OFF_BP);

    const int segBase = (blockIdx.x * 2 + wave) * 16;   // 16 segments per wave

    for (int si = 0; si < 16; si += 2) {                // process segments in pairs
#pragma unroll
        for (int p = 0; p < 2; ++p) {
            const size_t seg = (size_t)segBase + si + p;
#pragma unroll
            for (int q = 0; q < 3; ++q) {
                int t = q * 32 + lane;                  // 0..95 = [L=16][C=6]
                L.xbuf[p][6 + t] = imu[seg * 96 + t];
            }
        }
        wmma_layer<1>(L, A1, B1, nullptr, 0, si, lane);   // conv1:   6 ->  64
        wmma_layer<2>(L, A2, B2, nullptr, 0, si, lane);   // conv2:  64 -> 128
        wmma_layer<3>(L, A3, B3, nullptr, 0, si, lane);   // conv3: 128 -> 256 + mean pool
    }
    // projection: [256x256] x [256 x 16 segs] -> global out
    wmma_layer<4>(L, AP, BP, out, segBase, 0, lane);
}

// ================= launch =================
extern "C" void kernel_launch(void* const* d_in, const int* in_sizes, int n_in,
                              void* d_out, int out_size, void* d_ws, size_t ws_size,
                              hipStream_t stream)
{
    const float* imu = (const float*)d_in[0];
    const float* w1  = (const float*)d_in[1];
    const float* b1  = (const float*)d_in[2];
    const float* g1  = (const float*)d_in[3];
    const float* be1 = (const float*)d_in[4];
    const float* m1  = (const float*)d_in[5];
    const float* v1  = (const float*)d_in[6];
    const float* w2  = (const float*)d_in[7];
    const float* b2  = (const float*)d_in[8];
    const float* g2  = (const float*)d_in[9];
    const float* be2 = (const float*)d_in[10];
    const float* m2  = (const float*)d_in[11];
    const float* v2  = (const float*)d_in[12];
    const float* w3  = (const float*)d_in[13];
    const float* b3  = (const float*)d_in[14];
    const float* g3  = (const float*)d_in[15];
    const float* be3 = (const float*)d_in[16];
    const float* m3  = (const float*)d_in[17];
    const float* v3  = (const float*)d_in[18];
    const float* wp  = (const float*)d_in[19];
    const float* bp  = (const float*)d_in[20];

    unsigned int* wsU = (unsigned int*)d_ws;
    float*        wsF = (float*)d_ws;

    // weight / bias fragment prep (recomputed every call; deterministic)
    prep_A<<<4,   256, 0, stream>>>(w1, g1, v1, wsU + OFF_A1,   6,  4,  1, 0, 1024);
    prep_A<<<48,  256, 0, stream>>>(w2, g2, v2, wsU + OFF_A2,  64,  8,  6, 0, 12288);
    prep_A<<<192, 256, 0, stream>>>(w3, g3, v3, wsU + OFF_A3, 128, 16, 12, 0, 49152);
    prep_A<<<128, 256, 0, stream>>>(wp, nullptr, nullptr, wsU + OFF_AP, 256, 16, 8, 1, 32768);
    prep_bias<<<4,  256, 0, stream>>>(b1, g1, be1, m1, v1, wsF + OFF_B1, 0, 1024);
    prep_bias<<<8,  256, 0, stream>>>(b2, g2, be2, m2, v2, wsF + OFF_B2, 0, 2048);
    prep_bias<<<16, 256, 0, stream>>>(b3, g3, be3, m3, v3, wsF + OFF_B3, 0, 4096);
    prep_bias<<<16, 256, 0, stream>>>(bp, nullptr, nullptr, nullptr, nullptr, wsF + OFF_BP, 1, 4096);

    // 16384 segments / (2 waves * 16 segs) = 512 blocks of 64 threads
    imu_encoder_main<<<512, 64, 0, stream>>>(imu, wsU, (float*)d_out);
}
